// TemporalRunnerGNN_12893491823124
// MI455X (gfx1250) — matrix operand order repaired and checked by reading, not compiled, not run on hardware
//
#include <hip/hip_runtime.h>
#include <hip/hip_bf16.h>

// ---------------- problem constants ----------------
#define NODES   23
#define KNBR    8
#define BATCH   64
#define TSTEPS  128
#define GRAPHS  (BATCH * TSTEPS)          // 8192
#define NTOT    (GRAPHS * NODES)          // 188416
#define DIN     16
#define HDIM    128
#define RHDIM   128
#define EPERG   (NODES * KNBR)            // 184

typedef __attribute__((ext_vector_type(16))) __bf16 v16bf;
typedef __attribute__((ext_vector_type(8)))  __bf16 v8bf;
typedef __attribute__((ext_vector_type(8)))  float  v8f;

// ---------------- helpers ----------------
static __device__ inline __bf16 f2bf(float f) { return (__bf16)f; }   // native cvt, RNE
static __device__ inline float bf2f(__bf16 b) {
    unsigned u = ((unsigned)__builtin_bit_cast(unsigned short, b)) << 16;
    return __builtin_bit_cast(float, u);
}
static __device__ inline float fast_sigmoid(float x) {
    return __builtin_amdgcn_rcpf(1.0f + __expf(-x));
}
static __device__ inline float fast_tanh(float x) {
    return 1.0f - 2.0f * __builtin_amdgcn_rcpf(__expf(2.0f * x) + 1.0f);
}

// =============================================================================
// Pre-pack a B matrix (f32) into WMMA bf16 fragment order.
// frag = ntile*KT + kt; element (lane,e) = B[kt*32 + (e>>3)*16 + (lane>>4)*8 + (e&7)]
//                                           [ntile*16 + (lane&15)]  (0 if k>=K)
// TR=true: src is [N,K], B = src^T (W_ih / W_hh).
// =============================================================================
template<int Kdim, int Ncols, bool TR>
__global__ void prepack_b(const float* __restrict__ src, __bf16* __restrict__ dst) {
    constexpr int KT = (Kdim + 31) / 32;
    constexpr int TOTAL = (Ncols / 16) * KT * 32 * 16;
    int i = blockIdx.x * blockDim.x + threadIdx.x;
    if (i >= TOTAL) return;
    int e = i & 15;
    int l = (i >> 4) & 31;
    int frag = i >> 9;
    int kt = frag % KT, ntile = frag / KT;
    int k = kt * 32 + (e >> 3) * 16 + (l >> 4) * 8 + (e & 7);
    int n = ntile * 16 + (l & 15);
    float v = 0.0f;
    if (k < Kdim) v = TR ? src[(size_t)n * Kdim + k] : src[(size_t)k * Ncols + n];
    dst[i] = f2bf(v);
}

// =============================================================================
// WMMA bf16 GEMM: C[M,Ncols] = A_f32[M,Kdim] * Bfrag.  One wave owns a 16-row
// strip; 8 column-tiles at a time reuse each A fragment for 8 v_wmma ops.
// All shape logic compile-time; EXEC stays all-ones.
// =============================================================================
template<int Kdim, int Ncols>
__global__ void gemm_bf16_wmma(const float* __restrict__ A,
                               const __bf16* __restrict__ Bfrag,
                               float* __restrict__ C) {
    constexpr int KT = (Kdim + 31) / 32;
    constexpr int NTILES = Ncols / 16;
    const int wid  = blockIdx.x * (blockDim.x >> 5) + (threadIdx.x >> 5);
    const int l    = threadIdx.x & 31;
    const int nloc = l & 15;
    const int half = l >> 4;
    const int row  = wid * 16 + nloc;

    const float* arow = A + (size_t)row * Kdim;

    for (int nc = 0; nc < NTILES; nc += 8) {
        v8f acc[8] = {};
#pragma unroll
        for (int kt = 0; kt < KT; ++kt) {
            const int kbase = kt * 32;
            v16bf a;
            {   // k = kbase + half*8 + 0..7
                const float4* p = (const float4*)(arow + kbase + half * 8);
                float4 c0 = p[0], c1 = p[1];
                a[0] = f2bf(c0.x); a[1] = f2bf(c0.y); a[2] = f2bf(c0.z); a[3] = f2bf(c0.w);
                a[4] = f2bf(c1.x); a[5] = f2bf(c1.y); a[6] = f2bf(c1.z); a[7] = f2bf(c1.w);
            }
            if constexpr (Kdim >= 32) {   // k = kbase + 16 + half*8 + 0..7
                const float4* p = (const float4*)(arow + kbase + 16 + half * 8);
                float4 c0 = p[0], c1 = p[1];
                a[8]  = f2bf(c0.x); a[9]  = f2bf(c0.y); a[10] = f2bf(c0.z); a[11] = f2bf(c0.w);
                a[12] = f2bf(c1.x); a[13] = f2bf(c1.y); a[14] = f2bf(c1.z); a[15] = f2bf(c1.w);
            } else {                      // K=16: upper half of K-tile zero
#pragma unroll
                for (int e = 8; e < 16; ++e) a[e] = f2bf(0.0f);
            }
#pragma unroll
            for (int i = 0; i < 8; ++i) {
                const int ntile = nc + i;
                v16bf b = *(const v16bf*)(Bfrag + ((size_t)(ntile * KT + kt) * 32 + l) * 16);
                acc[i] = __builtin_amdgcn_wmma_f32_16x16x32_bf16(false, a, false, b,
                                                                 (short)0, acc[i], false, false);
            }
        }
#pragma unroll
        for (int i = 0; i < 8; ++i)
#pragma unroll
            for (int r = 0; r < 8; ++r)
                C[(size_t)(wid * 16 + r + 8 * half) * Ncols + (nc + i) * 16 + nloc] = acc[i][r];
    }
}

// =============================================================================
// GCN aggregation: deg==9 everywhere -> out = relu((self + 8 gathered)/9 + b)
// One wave per node, float4 per lane. Pure gather, no atomics.
// =============================================================================
__global__ void aggregate_all(const float* __restrict__ xw,
                              const int* __restrict__ src,
                              const float* __restrict__ bias,
                              float* __restrict__ hout) {
    const int node = blockIdx.x * (blockDim.x >> 5) + (threadIdx.x >> 5);
    const int lane = threadIdx.x & 31;
    const int g = node / NODES, j = node - g * NODES;
    const int ebase = g * EPERG + j * KNBR;

    float4 acc = reinterpret_cast<const float4*>(xw + (size_t)node * HDIM)[lane];
#pragma unroll
    for (int e = 0; e < KNBR; ++e) {
        int s = src[ebase + e];
        float4 v = reinterpret_cast<const float4*>(xw + (size_t)s * HDIM)[lane];
        acc.x += v.x; acc.y += v.y; acc.z += v.z; acc.w += v.w;
    }
    const float inv9 = 1.0f / 9.0f;
    float4 bb = reinterpret_cast<const float4*>(bias)[lane];
    float4 o;
    o.x = fmaxf(acc.x * inv9 + bb.x, 0.0f);
    o.y = fmaxf(acc.y * inv9 + bb.y, 0.0f);
    o.z = fmaxf(acc.z * inv9 + bb.z, 0.0f);
    o.w = fmaxf(acc.w * inv9 + bb.w, 0.0f);
    reinterpret_cast<float4*>(hout + (size_t)node * HDIM)[lane] = o;
}

__global__ void aggregate_runner(const float* __restrict__ hw,
                                 const int* __restrict__ src,
                                 const int* __restrict__ runner,
                                 const float* __restrict__ bias,
                                 float* __restrict__ seq) {
    const int g = blockIdx.x * (blockDim.x >> 5) + (threadIdx.x >> 5);
    const int lane = threadIdx.x & 31;
    const int r = runner[g];
    const int self = g * NODES + r;
    const int ebase = g * EPERG + r * KNBR;

    float4 acc = reinterpret_cast<const float4*>(hw + (size_t)self * HDIM)[lane];
#pragma unroll
    for (int e = 0; e < KNBR; ++e) {
        int s = src[ebase + e];
        float4 v = reinterpret_cast<const float4*>(hw + (size_t)s * HDIM)[lane];
        acc.x += v.x; acc.y += v.y; acc.z += v.z; acc.w += v.w;
    }
    const float inv9 = 1.0f / 9.0f;
    float4 bb = reinterpret_cast<const float4*>(bias)[lane];
    float4 o;
    o.x = fmaxf(acc.x * inv9 + bb.x, 0.0f);
    o.y = fmaxf(acc.y * inv9 + bb.y, 0.0f);
    o.z = fmaxf(acc.z * inv9 + bb.z, 0.0f);
    o.w = fmaxf(acc.w * inv9 + bb.w, 0.0f);
    reinterpret_cast<float4*>(seq + (size_t)g * HDIM)[lane] = o;
}

// =============================================================================
// Persistent GRU + fused projection. 512 threads = 16 waves = 4 waves/SIMD.
// waves_per_eu(4,4) => 256-VGPR budget per wave: the 12 loop-invariant B
// fragments (96 VGPRs) + 6 accumulators (48) stay register-resident, zero
// spills. Each wave owns TWO m-tiles sharing those B fragments: per k-tile
// 2 LDS A-loads + 6 back-to-back v_wmma.
// =============================================================================
__global__ void
__launch_bounds__(512, 1)
__attribute__((amdgpu_waves_per_eu(4, 4)))
gru_project(const float* __restrict__ xg,       // [G,384]
            const __bf16* __restrict__ whhF,    // prepacked frags
            const float* __restrict__ b_ih,
            const float* __restrict__ b_hh,
            const float* __restrict__ Wp,       // [128,2]
            const float* __restrict__ bp,
            float* __restrict__ out) {          // [B,T,2]
    __shared__ __bf16 hb[2][BATCH][RHDIM];      // 32 KB ping-pong hidden state
    __shared__ float  wpS[RHDIM * 2 + 2];

    const int tid   = threadIdx.x;
    const int l     = tid & 31;
    const int w     = tid >> 5;          // 0..15
    const int mpair = w >> 3;            // 0..1  -> m-tiles {2*mpair, 2*mpair+1}
    const int ctile = w & 7;             // 0..7  (hidden cols)
    const int nloc  = l & 15;
    const int half  = l >> 4;

    for (int i = tid; i < 2 * BATCH * RHDIM; i += blockDim.x)
        ((__bf16*)hb)[i] = f2bf(0.0f);
    if (tid < RHDIM * 2) wpS[tid] = Wp[tid];
    if (tid < 2)         wpS[RHDIM * 2 + tid] = bp[tid];
    __syncthreads();

    const int j = ctile * 16 + nloc;
    const float bihr = b_ih[j],             bhhr = b_hh[j];
    const float bihz = b_ih[RHDIM + j],     bhhz = b_hh[RHDIM + j];
    const float bihn = b_ih[2 * RHDIM + j], bhhn = b_hh[2 * RHDIM + j];

    int p = 0;
    for (int t = 0; t < TSTEPS; ++t) {
        v8f cr[2] = {}, cz[2] = {}, cn[2] = {};
#pragma unroll
        for (int kt = 0; kt < 4; ++kt) {
            const int kbase = kt * 32;
            // B fragments loop-invariant over t -> hoisted into VGPRs
            v16bf br = *(const v16bf*)(whhF + ((size_t)((ctile)      * 4 + kt) * 32 + l) * 16);
            v16bf bz = *(const v16bf*)(whhF + ((size_t)((8 + ctile)  * 4 + kt) * 32 + l) * 16);
            v16bf bn = *(const v16bf*)(whhF + ((size_t)((16 + ctile) * 4 + kt) * 32 + l) * 16);
#pragma unroll
            for (int m = 0; m < 2; ++m) {
                const int mtile = mpair * 2 + m;
                const __bf16* hrow = &hb[p][mtile * 16 + nloc][0];
                v16bf a;
                v8bf c0 = *(const v8bf*)(hrow + kbase + half * 8);
                v8bf c1 = *(const v8bf*)(hrow + kbase + 16 + half * 8);
#pragma unroll
                for (int e = 0; e < 8; ++e) { a[e] = c0[e]; a[8 + e] = c1[e]; }
                cr[m] = __builtin_amdgcn_wmma_f32_16x16x32_bf16(false, a, false, br, (short)0, cr[m], false, false);
                cz[m] = __builtin_amdgcn_wmma_f32_16x16x32_bf16(false, a, false, bz, (short)0, cz[m], false, false);
                cn[m] = __builtin_amdgcn_wmma_f32_16x16x32_bf16(false, a, false, bn, (short)0, cn[m], false, false);
            }
        }
        __syncthreads();   // all A-fragment reads of hb[p] complete

#pragma unroll
        for (int m = 0; m < 2; ++m) {
            const int mtile = mpair * 2 + m;
#pragma unroll
            for (int r8 = 0; r8 < 8; ++r8) {
                const int bat = mtile * 16 + r8 + 8 * half;
                const float* xr = xg + ((size_t)bat * TSTEPS + t) * (3 * RHDIM);
                float rr = fast_sigmoid(xr[j]             + bihr + cr[m][r8] + bhhr);
                float zz = fast_sigmoid(xr[RHDIM + j]     + bihz + cz[m][r8] + bhhz);
                float nn = fast_tanh   (xr[2 * RHDIM + j] + bihn + rr * (cn[m][r8] + bhhn));
                float hp = bf2f(hb[p][bat][j]);
                hb[p ^ 1][bat][j] = f2bf((1.0f - zz) * nn + zz * hp);
            }
        }
        __syncthreads();   // h_t fully written

        if (tid < BATCH * 2) {           // fused projection -> out[b,t,0:2]
            const int bat = tid >> 1, cc = tid & 1;
            float s = wpS[RHDIM * 2 + cc];
#pragma unroll
            for (int k = 0; k < RHDIM; ++k)
                s += bf2f(hb[p ^ 1][bat][k]) * wpS[k * 2 + cc];
            out[((size_t)bat * TSTEPS + t) * 2 + cc] = s;
        }
        p ^= 1;
    }
}

// ---------------- host-side launch ------------------------------------------
extern "C" void kernel_launch(void* const* d_in, const int* in_sizes, int n_in,
                              void* d_out, int out_size, void* d_ws, size_t ws_size,
                              hipStream_t stream) {
    const float* x      = (const float*)d_in[0];
    const int*   eidx   = (const int*)  d_in[1];   // [2, EDGES]; row 0 = src
    const int*   runner = (const int*)  d_in[2];
    const float* W1     = (const float*)d_in[3];
    const float* b1     = (const float*)d_in[4];
    const float* W2     = (const float*)d_in[5];
    const float* b2     = (const float*)d_in[6];
    const float* Wih    = (const float*)d_in[7];
    const float* Whh    = (const float*)d_in[8];
    const float* b_ih   = (const float*)d_in[9];
    const float* b_hh   = (const float*)d_in[10];
    const float* Wp     = (const float*)d_in[11];
    const float* bp     = (const float*)d_in[12];
    float* out = (float*)d_out;
    char*  ws  = (char*)d_ws;

    const size_t n_w1f  = (HDIM / 16) * 1 * 512;          // K=16 -> KT=1
    const size_t n_w2f  = (HDIM / 16) * 4 * 512;          // K=128 -> KT=4
    const size_t n_wihf = (3 * RHDIM / 16) * 4 * 512;
    const size_t n_whhf = (3 * RHDIM / 16) * 4 * 512;

    size_t off = 0;
    __bf16* w1f  = (__bf16*)(ws + off); off += (n_w1f  * 2 + 255) & ~255ull;
    __bf16* w2f  = (__bf16*)(ws + off); off += (n_w2f  * 2 + 255) & ~255ull;
    __bf16* wihf = (__bf16*)(ws + off); off += (n_wihf * 2 + 255) & ~255ull;
    __bf16* whhf = (__bf16*)(ws + off); off += (n_whhf * 2 + 255) & ~255ull;
    float* xw  = (float*)(ws + off); off += (size_t)NTOT * HDIM * 4;   // reused as hw
    float* h1  = (float*)(ws + off); off += (size_t)NTOT * HDIM * 4;
    float* seq = (float*)(ws + off); off += (size_t)GRAPHS * HDIM * 4;
    float* xg  = (float*)(ws + off); off += (size_t)GRAPHS * 3 * RHDIM * 4;

    // 0) pre-pack weights into WMMA fragment order (bf16)
    prepack_b<DIN,  HDIM,      false><<<(n_w1f  + 255) / 256, 256, 0, stream>>>(W1,  w1f);
    prepack_b<HDIM, HDIM,      false><<<(n_w2f  + 255) / 256, 256, 0, stream>>>(W2,  w2f);
    prepack_b<HDIM, 3 * RHDIM, true ><<<(n_wihf + 255) / 256, 256, 0, stream>>>(Wih, wihf);
    prepack_b<HDIM, 3 * RHDIM, true ><<<(n_whhf + 255) / 256, 256, 0, stream>>>(Whh, whhf);

    // 1) xw = x @ W1            [N,16]x[16,128]
    gemm_bf16_wmma<DIN, HDIM><<<(NTOT / 16) / 4, 128, 0, stream>>>(x, w1f, xw);
    // 2) h1 = relu(agg(xw)/9 + b1)
    aggregate_all<<<NTOT / 8, 256, 0, stream>>>(xw, eidx, b1, h1);
    // 3) hw = h1 @ W2           [N,128]x[128,128] (reuse xw)
    gemm_bf16_wmma<HDIM, HDIM><<<(NTOT / 16) / 4, 128, 0, stream>>>(h1, w2f, xw);
    // 4) seq[g] = relu(agg at runner /9 + b2)
    aggregate_runner<<<GRAPHS / 8, 256, 0, stream>>>(xw, eidx, runner, b2, seq);
    // 5) xg = seq @ W_ih^T      [8192,128]x[128,384]
    gemm_bf16_wmma<HDIM, 3 * RHDIM><<<(GRAPHS / 16) / 4, 128, 0, stream>>>(seq, wihf, xg);
    // 6) GRU recurrence + output projection (16 waves, 256 VGPRs/wave)
    gru_project<<<1, 512, 0, stream>>>(xg, whhf, b_ih, b_hh, Wp, bp, out);
}
